// QuantumLikeLayer_21251498181499
// MI455X (gfx1250) — matrix operand order repaired and checked by reading, not compiled, not run on hardware
//
#include <hip/hip_runtime.h>
#include <cmath>

typedef __attribute__((ext_vector_type(16))) _Float16 v16h;
typedef __attribute__((ext_vector_type(8)))  float    v8f;
typedef __attribute__((ext_vector_type(8)))  float    f32x8;
typedef __attribute__((ext_vector_type(2)))  float    f32x2;

#define IN_F   512
#define NQ     6
#define NL     12
#define BATCH  65536
#define PI_F   3.14159265358979f

#define BLOCK  256
#define WAVES  (BLOCK / 32)
#define ROWS_PER_BLOCK (WAVES * 16)

__global__ __launch_bounds__(BLOCK)
void qlayer_kernel(const float* __restrict__ x,
                   const float* __restrict__ W,
                   const float* __restrict__ b,
                   const float* __restrict__ theta,
                   float* __restrict__ out)
{
    // W^T staged as f16, N padded 6 -> 16 (zeros): 16 KB
    __shared__ __align__(32) _Float16 Wt[16][IN_F];
    // per-wave 16 rows x (6 used of 8) post-tanh h values: 4 KB
    __shared__ float hbuf[WAVES][16][8];
    __shared__ float ths[NL * NQ * 3];
    __shared__ float bs[8];

    const int tid = threadIdx.x;

    // ---- stage W^T (f16, zero-padded) + theta + bias into LDS ----
    for (int i = tid; i < 16 * IN_F; i += BLOCK) {
        int n = i >> 9;        // / 512
        int k = i & (IN_F - 1);
        Wt[n][k] = (n < NQ) ? (_Float16)W[k * NQ + n] : (_Float16)0.0f;
    }
    for (int i = tid; i < NL * NQ * 3; i += BLOCK) ths[i] = theta[i];
    if (tid < 8) bs[tid] = (tid < NQ) ? b[tid] : 0.0f;
    __syncthreads();

    const int wid  = tid >> 5;
    const int lane = tid & 31;
    const int half = lane >> 4;    // hi/lo half of wave32
    const int lidx = lane & 15;    // M index for A rows, N index for B/C cols
    const int m0   = blockIdx.x * ROWS_PER_BLOCK + wid * 16;

    // ---- GEMM: C[16x16] = x_tile[16x512] * W^T_padded  (f16 in, f32 acc) ----
    v8f c = {};
    const float* xrow = x + (size_t)(m0 + lidx) * IN_F;

    #pragma unroll
    for (int k0 = 0; k0 < IN_F; k0 += 32) {
        // A fragment: lane(M=lidx, half) needs K = k0+half*16..+15
        f32x8 r0 = *(const f32x8*)(xrow + k0 + half * 16);
        f32x8 r1 = *(const f32x8*)(xrow + k0 + half * 16 + 8);
        v16h a;
        #pragma unroll
        for (int e = 0; e < 8; ++e) {
            a[e]     = (_Float16)r0[e];
            a[8 + e] = (_Float16)r1[e];
        }
        // B fragment: lane(N=lidx, half) holds K = k0 + half*16 + (0..15), contiguous in Wt row
        v16h bf = *(const v16h*)&Wt[lidx][k0 + half * 16];

        c = __builtin_amdgcn_wmma_f32_16x16x32_f16(
                /*neg_a=*/false, a, /*neg_b=*/false, bf,
                /*c_mod=*/(short)0, c, /*reuse_a=*/false, /*reuse_b=*/false);
    }

    // ---- epilogue: h = tanh(c + bias) * pi, stage 16x6 per wave to LDS ----
    {
        const int   n    = lidx;             // C column (N) held by this lane
        const float bias = bs[n & 7];
        if (n < 8) {
            #pragma unroll
            for (int r = 0; r < 8; ++r) {
                // C layout: VGPR r holds M = r (lanes 0-15) / M = r+8 (lanes 16-31)
                hbuf[wid][r + 8 * half][n] = tanhf(c[r] + bias) * PI_F;
            }
        }
    }
    __syncthreads();

    // ---- recurrence: lanes 0..15 of each wave handle one row each ----
    if (lane < 16) {
        const int row = m0 + lane;
        float hn[NQ];
        #pragma unroll
        for (int n = 0; n < NQ; ++n) hn[n] = hbuf[wid][lane][n];

        // layer 0: h differs per qubit
        float s = 0.0f;
        #pragma unroll
        for (int n = 0; n < NQ; ++n) {
            float t0 = ths[n * 3 + 0];
            float t1 = ths[n * 3 + 1];
            float t2 = ths[n * 3 + 2];
            s += __sinf(hn[n] + t0) + __cosf(hn[n] + t1) + __sinf(hn[n] * t2);
        }
        float h = s * (1.0f / NQ);

        // layers 1..11: h uniform across qubits (mean broadcast)
        #pragma unroll
        for (int l = 1; l < NL; ++l) {
            float s2 = 0.0f;
            #pragma unroll
            for (int n = 0; n < NQ; ++n) {
                float t0 = ths[(l * NQ + n) * 3 + 0];
                float t1 = ths[(l * NQ + n) * 3 + 1];
                float t2 = ths[(l * NQ + n) * 3 + 2];
                s2 += __sinf(h + t0) + __cosf(h + t1) + __sinf(h * t2);
            }
            h = s2 * (1.0f / NQ);
        }

        // row stride = 24 B -> 8 B aligned; use three float2 stores
        float* o = out + (size_t)row * NQ;
        f32x2 v2 = {h, h};
        *(f32x2*)(o + 0) = v2;
        *(f32x2*)(o + 2) = v2;
        *(f32x2*)(o + 4) = v2;
    }
}

extern "C" void kernel_launch(void* const* d_in, const int* in_sizes, int n_in,
                              void* d_out, int out_size, void* d_ws, size_t ws_size,
                              hipStream_t stream) {
    const float* x     = (const float*)d_in[0];
    const float* W     = (const float*)d_in[1];
    const float* b     = (const float*)d_in[2];
    const float* theta = (const float*)d_in[3];
    float* out = (float*)d_out;

    dim3 grid(BATCH / ROWS_PER_BLOCK);   // 512 blocks of 256 threads (8 waves)
    qlayer_kernel<<<grid, BLOCK, 0, stream>>>(x, W, b, theta, out);
}
